// NonMaximumSuppression_67259187856050
// MI455X (gfx1250) — compile-verified, matrix operand order checked by reading, not run on hardware
//
#include <hip/hip_runtime.h>
#include <hip/hip_bf16.h>
#include <hip/hip_fp16.h>

#define N_ANCH 300000
#define N_CLS 80
#define DETD 85
#define TOPK 1000
#define PADK 1024
#define MAXB 300
#define NMS_THR 0.4f

typedef _Float16 v16h __attribute__((ext_vector_type(16)));
typedef float    v8f  __attribute__((ext_vector_type(8)));

// ---------------------------------------------------------------------------
// Pass 1: scores[i] = max over 80 classes. 96MB stream -> HBM bound (~4.1us).
// ---------------------------------------------------------------------------
__global__ void k_scores(const float* __restrict__ cls, float* __restrict__ scores, int n) {
    int i = blockIdx.x * blockDim.x + threadIdx.x;
    if (i >= n) return;
    const float4* p = reinterpret_cast<const float4*>(cls + (size_t)i * N_CLS);
    __builtin_prefetch(p + (N_CLS / 4), 0, 0);   // global_prefetch_b8 (next row)
    float m = -1.0f;
#pragma unroll
    for (int j = 0; j < N_CLS / 4; ++j) {
        float4 v = p[j];
        m = fmaxf(m, fmaxf(fmaxf(v.x, v.y), fmaxf(v.z, v.w)));
    }
    scores[i] = m;
}

// ---------------------------------------------------------------------------
// Radix top-k select (4 byte rounds). state = {prefix, kRem, countAbove}
// Scores are non-negative floats -> uint bit compare == float compare.
// ---------------------------------------------------------------------------
__global__ void k_state_init(unsigned* state) {
    if (threadIdx.x == 0) { state[0] = 0u; state[1] = TOPK; state[2] = 0u; }
}

__global__ void k_hist_clear(unsigned* hist) { hist[threadIdx.x] = 0u; }

__global__ void k_hist(const float* __restrict__ scores, const unsigned* __restrict__ state,
                       unsigned* __restrict__ hist, int shift, int n) {
    __shared__ unsigned lh[256];
    int t = threadIdx.x;
    lh[t] = 0u;
    __syncthreads();
    unsigned prefix = state[0];
    unsigned mask = (shift == 24) ? 0u : (0xFFFFFFFFu << (shift + 8));
    int i = blockIdx.x * blockDim.x + t;
    if (i < n) {
        unsigned bits = __float_as_uint(scores[i]);
        if ((bits & mask) == prefix)
            atomicAdd(&lh[(bits >> shift) & 0xFFu], 1u);
    }
    __syncthreads();
    if (lh[t]) atomicAdd(&hist[t], lh[t]);
}

__global__ void k_select(unsigned* state, const unsigned* hist, int shift) {
    if (threadIdx.x != 0) return;
    unsigned prefix = state[0], kRem = state[1], cA = state[2];
    for (int d = 255; d >= 0; --d) {
        unsigned c = hist[d];
        if (c >= kRem) { prefix |= ((unsigned)d) << shift; break; }
        kRem -= c; cA += c;
    }
    state[0] = prefix; state[1] = kRem; state[2] = cA;
}

__global__ void k_cand_init(float* cscore, unsigned* cidx, unsigned* cnts) {
    int i = blockIdx.x * blockDim.x + threadIdx.x;
    if (i < PADK) { cscore[i] = -1.0f; cidx[i] = 0u; }
    if (i < 2) cnts[i] = 0u;
}

__global__ void k_zero_out(float* out, int n) {
    int i = blockIdx.x * blockDim.x + threadIdx.x;
    if (i < n) out[i] = 0.0f;
}

__global__ void k_compact(const float* __restrict__ scores, const unsigned* __restrict__ state,
                          unsigned* cnts, float* cscore, unsigned* cidx, int n) {
    int i = blockIdx.x * blockDim.x + threadIdx.x;
    if (i >= n) return;
    float s = scores[i];
    unsigned bits = __float_as_uint(s);
    unsigned T = state[0], kRem = state[1], cA = state[2];
    int slot = -1;
    if (bits > T) {
        unsigned p = atomicAdd(&cnts[0], 1u);
        if (p < PADK) slot = (int)p;                 // p < cA <= 999 by construction
    } else if (bits == T) {
        unsigned t = atomicAdd(&cnts[1], 1u);
        if (t < kRem) slot = (int)(cA + t);          // fills exactly to TOPK
    }
    if (slot >= 0 && slot < PADK) { cscore[slot] = s; cidx[slot] = (unsigned)i; }
}

// ---------------------------------------------------------------------------
// Bitonic sort 1024 (score desc, idx payload) in LDS; then gather top-k boxes.
// Pads (slots >= TOPK) get distant non-overlapping boxes.
// ---------------------------------------------------------------------------
__global__ void k_sort_gather(float* cscore, unsigned* cidx,
                              const float* __restrict__ boxes, float* __restrict__ boxes_k) {
    __shared__ float    s[PADK];
    __shared__ unsigned id[PADK];
    int t = threadIdx.x;                       // 512 threads
    s[t] = cscore[t];           id[t] = cidx[t];
    s[t + 512] = cscore[t + 512]; id[t + 512] = cidx[t + 512];
    for (unsigned k = 2; k <= PADK; k <<= 1) {
        for (unsigned j = k >> 1; j > 0; j >>= 1) {
            __syncthreads();
            for (unsigned tid = t; tid < PADK; tid += 512) {
                unsigned ixj = tid ^ j;
                if (ixj > tid) {
                    bool desc = ((tid & k) == 0);
                    bool sw = desc ? (s[tid] < s[ixj]) : (s[tid] > s[ixj]);
                    if (sw) {
                        float fs = s[tid]; s[tid] = s[ixj]; s[ixj] = fs;
                        unsigned fi = id[tid]; id[tid] = id[ixj]; id[ixj] = fi;
                    }
                }
            }
        }
    }
    __syncthreads();
    for (unsigned e = t; e < PADK; e += 512) {
        cscore[e] = s[e]; cidx[e] = id[e];
        if (e < TOPK) {
            const float* b = boxes + (size_t)id[e] * 4;
            boxes_k[e * 4 + 0] = b[0]; boxes_k[e * 4 + 1] = b[1];
            boxes_k[e * 4 + 2] = b[2]; boxes_k[e * 4 + 3] = b[3];
        } else {
            float base = 1.0e7f + (float)e * 4.0f;  // isolated pad boxes, IoU==0 vs all
            boxes_k[e * 4 + 0] = base;        boxes_k[e * 4 + 1] = base;
            boxes_k[e * 4 + 2] = base + 0.5f; boxes_k[e * 4 + 3] = base + 0.5f;
        }
    }
}

// ---------------------------------------------------------------------------
// Overlap indicator matrix Ovl[r][c] = (IoU > 0.4) ? 1.0h : 0.0h  (1024x1024 f16)
// ---------------------------------------------------------------------------
__global__ void k_iou(const float* __restrict__ bk, _Float16* __restrict__ ovl) {
    int r = blockIdx.x;
    float y1 = bk[r * 4 + 0], x1 = bk[r * 4 + 1], y2 = bk[r * 4 + 2], x2 = bk[r * 4 + 3];
    float ar = (y2 - y1) * (x2 - x1);
    int c0 = threadIdx.x * 4;
#pragma unroll
    for (int q = 0; q < 4; ++q) {
        int c = c0 + q;
        float cy1 = bk[c * 4 + 0], cx1 = bk[c * 4 + 1], cy2 = bk[c * 4 + 2], cx2 = bk[c * 4 + 3];
        float ac = (cy2 - cy1) * (cx2 - cx1);
        float ih = fminf(y2, cy2) - fmaxf(y1, cy1); ih = fmaxf(ih, 0.0f);
        float iw = fminf(x2, cx2) - fmaxf(x1, cx1); iw = fmaxf(iw, 0.0f);
        float inter = ih * iw;
        float iou = inter / (ar + ac - inter + 1e-8f);
        ovl[(size_t)r * PADK + c] = (iou > NMS_THR) ? (_Float16)1.0f : (_Float16)0.0f;
    }
}

// ---------------------------------------------------------------------------
// Greedy NMS scan, WMMA-blocked. For block b (16 rows):
//   S[m] = sum_k Ovl[16b+m][k] * keep[k]   via v_wmma_f32_16x16x32_f16 chunks
// then 16-step serial pass on the diagonal tile. Exact (0/1 values in f16).
// Single wave32; EXEC all-ones at every WMMA (uniform control flow).
// ---------------------------------------------------------------------------
__global__ void k_scan(const _Float16* __restrict__ ovl, unsigned* __restrict__ keep) {
    __shared__ _Float16 keepf[PADK];
    __shared__ float    S[16];
    __shared__ _Float16 diag[256];
    __shared__ unsigned smask;
    int lane = threadIdx.x;                 // 32 threads
    for (int t = lane; t < PADK; t += 32) keepf[t] = (_Float16)0.0f;
    __syncthreads();

    int half = lane >> 4;
    int m = lane & 15;

    for (int b = 0; b < PADK / 16; ++b) {
        v8f acc = {};
        int nch = (b + 1) >> 1;             // 32-col chunks covering cols [0, 16b) (+own block, keepf=0)
        const _Float16* rowp = ovl + (size_t)(16 * b + m) * PADK;
        for (int c = 0; c < nch; ++c) {
            v16h A, B;
#pragma unroll
            for (int e = 0; e < 16; ++e) {
                int K = ((e < 8) ? e : (e + 8)) + 8 * half;  // ISA 16-bit A 16x32 layout
                A[e] = rowp[32 * c + K];
                B[e] = keepf[32 * c + K];                    // B[k][n] = keep[k] (all cols equal)
            }
            acc = __builtin_amdgcn_wmma_f32_16x16x32_f16(
                false, A, false, B, (short)0, acc, false, false);
        }
        // D layout: lanes 0-15 = N, VGPR r -> M=r (lanes 0-15) / M=8+r (lanes 16-31)
        if (lane == 0) {
#pragma unroll
            for (int r = 0; r < 8; ++r) S[r] = acc[r];
        }
        if (lane == 16) {
#pragma unroll
            for (int r = 0; r < 8; ++r) S[8 + r] = acc[r];
        }
        // diagonal 16x16 tile
        for (int t = lane; t < 256; t += 32)
            diag[t] = ovl[(size_t)(16 * b + (t >> 4)) * PADK + 16 * b + (t & 15)];
        __syncthreads();

        if (lane == 0) {
            unsigned km = 0;
            for (int mm = 0; mm < 16; ++mm) {
                bool sup = S[mm] > 0.5f;
                for (int j = 0; j < mm && !sup; ++j)
                    if (((km >> j) & 1u) && ((float)diag[mm * 16 + j] > 0.5f)) sup = true;
                if (!sup) km |= (1u << mm);
            }
            smask = km;
        }
        __syncthreads();
        if (lane < 16) {
            unsigned kb = (smask >> lane) & 1u;
            keepf[16 * b + lane] = kb ? (_Float16)1.0f : (_Float16)0.0f;
            keep[16 * b + lane] = kb;
        }
        __syncthreads();
    }
}

// ---------------------------------------------------------------------------
// Compaction: exclusive prefix over keep[0..999], copy first 300 kept det rows.
// ---------------------------------------------------------------------------
__global__ void k_emit(const unsigned* __restrict__ keep, const unsigned* __restrict__ cidx,
                       const float* __restrict__ det, float* __restrict__ out) {
    __shared__ int pre[PADK];
    int i = threadIdx.x;                   // 1024 threads
    int flag = (i < TOPK) ? (int)keep[i] : 0;
    pre[i] = flag;
    __syncthreads();
    for (int off = 1; off < PADK; off <<= 1) {
        int v = (i >= off) ? pre[i - off] : 0;
        __syncthreads();
        pre[i] += v;
        __syncthreads();
    }
    int pos = pre[i] - flag;               // exclusive rank
    if (flag && pos < MAXB) {
        const float* src = det + (size_t)cidx[i] * DETD;
        float* dst = out + (size_t)pos * DETD;
        for (int c = 0; c < DETD; ++c) dst[c] = src[c];
    }
}

// ---------------------------------------------------------------------------
extern "C" void kernel_launch(void* const* d_in, const int* in_sizes, int n_in,
                              void* d_out, int out_size, void* d_ws, size_t ws_size,
                              hipStream_t stream) {
    (void)in_sizes; (void)n_in; (void)out_size; (void)ws_size;
    const float* boxes = (const float*)d_in[0];
    const float* cls   = (const float*)d_in[1];
    const float* det   = (const float*)d_in[2];
    float* out = (float*)d_out;

    char* ws = (char*)d_ws;                      // layout (256B aligned regions):
    float*     scores  = (float*)(ws + 0);                       // 300000 f32
    unsigned*  state   = (unsigned*)(ws + 1200128);              // prefix,kRem,countAbove
    unsigned*  hist    = (unsigned*)(ws + 1200128 + 256);        // 256 u32
    unsigned*  cnts    = (unsigned*)(ws + 1200128 + 256 + 1024); // 2 u32
    float*     cscore  = (float*)(ws + 1201664);                 // 1024 f32
    unsigned*  cidx    = (unsigned*)(ws + 1205760);              // 1024 u32
    float*     boxes_k = (float*)(ws + 1209856);                 // 1024*4 f32
    unsigned*  keep    = (unsigned*)(ws + 1226240);              // 1024 u32
    _Float16*  ovl     = (_Float16*)(ws + 1230336);              // 1024*1024 f16 (2MB)

    int nb = (N_ANCH + 255) / 256;

    k_scores<<<nb, 256, 0, stream>>>(cls, scores, N_ANCH);
    k_state_init<<<1, 1, 0, stream>>>(state);
    for (int r = 0; r < 4; ++r) {
        int shift = 24 - 8 * r;
        k_hist_clear<<<1, 256, 0, stream>>>(hist);
        k_hist<<<nb, 256, 0, stream>>>(scores, state, hist, shift, N_ANCH);
        k_select<<<1, 1, 0, stream>>>(state, hist, shift);
    }
    k_cand_init<<<4, 256, 0, stream>>>(cscore, cidx, cnts);
    k_zero_out<<<(MAXB * DETD + 255) / 256, 256, 0, stream>>>(out, MAXB * DETD);
    k_compact<<<nb, 256, 0, stream>>>(scores, state, cnts, cscore, cidx, N_ANCH);
    k_sort_gather<<<1, 512, 0, stream>>>(cscore, cidx, boxes, boxes_k);
    k_iou<<<PADK, 256, 0, stream>>>(boxes_k, ovl);
    k_scan<<<1, 32, 0, stream>>>(ovl, keep);
    k_emit<<<1, 1024, 0, stream>>>(keep, cidx, det, out);
}
